// Net_31482110280058
// MI455X (gfx1250) — matrix operand order repaired and checked by reading, not compiled, not run on hardware
//
#include <hip/hip_runtime.h>

typedef __bf16 bf16_t;
typedef __bf16        v16bf __attribute__((ext_vector_type(16)));
typedef float         v8f   __attribute__((ext_vector_type(8)));
typedef unsigned int  v4u   __attribute__((ext_vector_type(4)));

union Frag { v16bf v; v4u u[2]; };

#define T_STEPS 200
#define BATCH   256
#define I_DIM   784
#define K1P     800      // 784 padded to multiple of 32
#define H_DIM   1000
#define H_PADN  1024     // padded weight rows so N-tile loads never go OOB
#define K2P     1024     // 1000 padded to multiple of 32
#define O_DIM   10
#define M_ROWS  (T_STEPS*BATCH)   // 51200

__device__ __forceinline__ v8f wmma_bf16(v16bf a, v16bf b, v8f c) {
  // (neg_a, A, neg_b, B, c_mod, C, reuse_a, reuse_b)
  return __builtin_amdgcn_wmma_f32_16x16x32_bf16(false, a, false, b,
                                                 (short)0, c, false, false);
}

// ---------------- conversion kernels (f32 -> bf16, zero-padded) -------------
__global__ void cvt_x(const float* __restrict__ x, bf16_t* __restrict__ Xb) {
  long i = (long)blockIdx.x * blockDim.x + threadIdx.x;   // over M_ROWS*K1P
  long m = i / K1P;
  int  k = (int)(i - m * K1P);
  Xb[i] = (k < I_DIM) ? (bf16_t)x[m * I_DIM + k] : (bf16_t)0.0f;
}

__global__ void cvt_w1(const float* __restrict__ w, bf16_t* __restrict__ Wb) {
  long i = (long)blockIdx.x * blockDim.x + threadIdx.x;   // over H_PADN*K1P
  long n = i / K1P;
  int  k = (int)(i - n * K1P);
  Wb[i] = (n < H_DIM && k < I_DIM) ? (bf16_t)w[n * I_DIM + k] : (bf16_t)0.0f;
}

__global__ void cvt_w2(const float* __restrict__ w, bf16_t* __restrict__ Wb) {
  int i = blockIdx.x * blockDim.x + threadIdx.x;          // over 16*K2P
  int n = i / K2P;
  int k = i - n * K2P;
  Wb[i] = (n < O_DIM && k < H_DIM) ? (bf16_t)w[n * H_DIM + k] : (bf16_t)0.0f;
}

// ---------------- GEMM1: Z1[51200,1000] = Xb @ W1b^T + bias -----------------
// Block = 8 waves arranged 2(M) x 4(N); each wave does 32x32 via 2x2 WMMA tiles.
__global__ __launch_bounds__(256) void gemm1(const bf16_t* __restrict__ Xb,
                                             const bf16_t* __restrict__ Wb,
                                             const float*  __restrict__ bih,
                                             const float*  __restrict__ bhh,
                                             float* __restrict__ Z) {
  const int lane = threadIdx.x & 31;
  const int wave = threadIdx.x >> 5;
  const int lm   = lane & 15;       // row/col within 16
  const int lh   = lane >> 4;       // half-wave select
  const long tileM = (long)blockIdx.x * 64 + (wave >> 2) * 32;
  const int  tileN = blockIdx.y * 128 + (wave & 3) * 32;

  const bf16_t* a0p = Xb + (tileM + lm)      * (long)K1P;
  const bf16_t* a1p = Xb + (tileM + 16 + lm) * (long)K1P;
  const bf16_t* b0p = Wb + (long)(tileN + lm)      * K1P;
  const bf16_t* b1p = Wb + (long)(tileN + 16 + lm) * K1P;

  v8f acc[2][2] = {};

  for (int k = 0; k < K1P; k += 32) {
    Frag a0, a1, b0, b1;
    // A 16x32 bf16 layout: lanes0-15 K=0..7 & 16..23; lanes16-31 K=8..15 & 24..31
    a0.u[0] = *(const v4u*)(a0p + k + lh * 8);
    a0.u[1] = *(const v4u*)(a0p + k + 16 + lh * 8);
    a1.u[0] = *(const v4u*)(a1p + k + lh * 8);
    a1.u[1] = *(const v4u*)(a1p + k + 16 + lh * 8);
    // B 32x16 bf16 layout: lanes0-15 K=0..7 & 16..23 of col lm; lanes16-31 K=8..15 & 24..31
    const bf16_t* pb0 = b0p + k + lh * 8;
    b0.u[0] = *(const v4u*)(pb0);
    b0.u[1] = *(const v4u*)(pb0 + 16);
    const bf16_t* pb1 = b1p + k + lh * 8;
    b1.u[0] = *(const v4u*)(pb1);
    b1.u[1] = *(const v4u*)(pb1 + 16);

    acc[0][0] = wmma_bf16(a0.v, b0.v, acc[0][0]);
    acc[0][1] = wmma_bf16(a0.v, b1.v, acc[0][1]);
    acc[1][0] = wmma_bf16(a1.v, b0.v, acc[1][0]);
    acc[1][1] = wmma_bf16(a1.v, b1.v, acc[1][1]);
  }

  // C/D layout: VGPR r, lane -> (M = r + 8*lh, N = lm)
  for (int nj = 0; nj < 2; ++nj) {
    int col = tileN + nj * 16 + lm;
    if (col >= H_DIM) continue;
    float bias = bih[col] + bhh[col];
    for (int mi = 0; mi < 2; ++mi) {
      long rowBase = tileM + mi * 16 + lh * 8;
      for (int r = 0; r < 8; ++r)
        Z[(rowBase + r) * H_DIM + col] = acc[mi][nj][r] + bias;
    }
  }
}

// ---------------- scan over T for layer 1; spikes emitted as bf16 -----------
__global__ void scan1(const float* __restrict__ Z, const float* __restrict__ beta,
                      bf16_t* __restrict__ S) {
  int j = blockIdx.x * blockDim.x + threadIdx.x;   // over BATCH*1024
  int b = j >> 10;
  int h = j & 1023;
  if (h >= H_DIM) {  // zero-fill K padding for GEMM2
    for (int t = 0; t < T_STEPS; ++t)
      S[((long)(t * BATCH + b) << 10) + h] = (bf16_t)0.0f;
    return;
  }
  float bt  = fminf(fmaxf(beta[h], 0.0f), 1.0f);
  float mem = 0.0f;
  for (int t = 0; t < T_STEPS; ++t) {
    float z = Z[(long)(t * BATCH + b) * H_DIM + h];
    mem = fmaxf(z + bt * mem, 0.0f);
    S[((long)(t * BATCH + b) << 10) + h] = (mem > 1.0f) ? (bf16_t)1.0f : (bf16_t)0.0f;
  }
}

// ---------------- GEMM2: Z2[51200,10] = S @ W2b^T + bias --------------------
__global__ __launch_bounds__(256) void gemm2(const bf16_t* __restrict__ S,
                                             const bf16_t* __restrict__ Wb,
                                             const float*  __restrict__ bih,
                                             const float*  __restrict__ bhh,
                                             float* __restrict__ Z2) {
  const int lane = threadIdx.x & 31;
  const int wave = threadIdx.x >> 5;
  const int lm   = lane & 15;
  const int lh   = lane >> 4;
  const long tileM = (long)blockIdx.x * 512 + wave * 64;

  const bf16_t* bp = Wb + (long)lm * K2P;  // column lm of B = row lm of W2b
  v8f acc[4] = {};
  for (int k = 0; k < K2P; k += 32) {
    Frag b;
    const bf16_t* pb = bp + k + lh * 8;
    b.u[0] = *(const v4u*)pb;
    b.u[1] = *(const v4u*)(pb + 16);
    for (int mi = 0; mi < 4; ++mi) {
      Frag a;
      const bf16_t* pa = S + (tileM + mi * 16 + lm) * (long)K2P + k + lh * 8;
      a.u[0] = *(const v4u*)pa;
      a.u[1] = *(const v4u*)(pa + 16);
      acc[mi] = wmma_bf16(a.v, b.v, acc[mi]);
    }
  }
  int col = lm;
  if (col < O_DIM) {
    float bias = bih[col] + bhh[col];
    for (int mi = 0; mi < 4; ++mi) {
      long rowBase = tileM + mi * 16 + lh * 8;
      for (int r = 0; r < 8; ++r)
        Z2[(rowBase + r) * O_DIM + col] = acc[mi][r] + bias;
    }
  }
}

// ---------------- scan over T for layer 2; f32 heaviside output -------------
__global__ void scan2(const float* __restrict__ Z2, const float* __restrict__ beta,
                      float* __restrict__ out) {
  int j = blockIdx.x * blockDim.x + threadIdx.x;
  if (j >= BATCH * O_DIM) return;
  int b = j / O_DIM;
  int o = j - b * O_DIM;
  float bt  = fminf(fmaxf(beta[o], 0.0f), 1.0f);
  float mem = 0.0f;
  for (int t = 0; t < T_STEPS; ++t) {
    float z = Z2[(long)(t * BATCH + b) * O_DIM + o];
    mem = fmaxf(z + bt * mem, 0.0f);
    out[(long)(t * BATCH + b) * O_DIM + o] = (mem > 1.0f) ? 1.0f : 0.0f;
  }
}

extern "C" void kernel_launch(void* const* d_in, const int* in_sizes, int n_in,
                              void* d_out, int out_size, void* d_ws, size_t ws_size,
                              hipStream_t stream) {
  const float* x     = (const float*)d_in[0];
  const float* W1    = (const float*)d_in[1];
  const float* bih1  = (const float*)d_in[2];
  const float* bhh1  = (const float*)d_in[3];
  const float* beta1 = (const float*)d_in[4];
  const float* W2    = (const float*)d_in[5];
  const float* bih2  = (const float*)d_in[6];
  const float* bhh2  = (const float*)d_in[7];
  const float* beta2 = (const float*)d_in[8];

  char* ws = (char*)d_ws;
  size_t off = 0;
  bf16_t* Xb  = (bf16_t*)(ws + off); off += (size_t)M_ROWS * K1P * 2;  // 81.9 MB
  bf16_t* W1b = (bf16_t*)(ws + off); off += (size_t)H_PADN * K1P * 2;  // 1.6 MB
  float*  Z1  = (float*) (ws + off); off += (size_t)M_ROWS * H_DIM * 4; // 204.8 MB
  bf16_t* S1  = (bf16_t*)(ws + off); off += (size_t)M_ROWS * K2P * 2;  // 104.9 MB
  bf16_t* W2b = (bf16_t*)(ws + off); off += (size_t)16 * K2P * 2;      // 32 KB
  float*  Z2  = (float*) (ws + off); off += (size_t)M_ROWS * O_DIM * 4; // 2.0 MB

  cvt_x <<<(M_ROWS * K1P) / 256, 256, 0, stream>>>(x, Xb);
  cvt_w1<<<(H_PADN * K1P) / 256, 256, 0, stream>>>(W1, W1b);
  cvt_w2<<<(16 * K2P) / 256, 256, 0, stream>>>(W2, W2b);

  gemm1<<<dim3(M_ROWS / 64, 8), 256, 0, stream>>>(Xb, W1b, bih1, bhh1, Z1);
  scan1<<<(BATCH * 1024) / 256, 256, 0, stream>>>(Z1, beta1, S1);
  gemm2<<<M_ROWS / 512, 256, 0, stream>>>(S1, W2b, bih2, bhh2, Z2);
  scan2<<<(BATCH * O_DIM + 255) / 256, 256, 0, stream>>>(Z2, beta2, (float*)d_out);
}
